// Injector_89386859364818
// MI455X (gfx1250) — matrix operand order repaired and checked by reading, not compiled, not run on hardware
//
#include <hip/hip_runtime.h>
#include <hip/hip_bf16.h>
#include <math.h>

typedef __bf16 bf16_t;
typedef __attribute__((ext_vector_type(16))) __bf16 v16bf;
typedef __attribute__((ext_vector_type(8)))  __bf16 v8bf;
typedef __attribute__((ext_vector_type(8)))  float  v8f;

#define E_DIM 768
#define NH_   8
#define HD_   96
#define NP_   4
#define NQ_   4096
#define NV_   9216
#define B_    4
#define HF_   96
#define WF_   96

// ---------------------------------------------------------------------------
// Transpose-cast fp32 W[K][N] -> bf16 Wt[N][K]
// ---------------------------------------------------------------------------
__global__ __launch_bounds__(256) void castT_kernel(const float* __restrict__ W,
                                                    bf16_t* __restrict__ Wt,
                                                    int Kdim, int Ncols) {
  int t = blockIdx.x * 256 + threadIdx.x;
  if (t >= Kdim * Ncols) return;
  int k = t / Ncols, n = t % Ncols;
  Wt[(size_t)n * Kdim + k] = (bf16_t)W[t];
}

// concat b_off(64) + b_attn(32) -> b_oa(96)
__global__ void boa_kernel(const float* b_off, const float* b_attn, float* b_oa) {
  int t = threadIdx.x;
  if (t < 96) b_oa[t] = (t < 64) ? b_off[t] : b_attn[t - 64];
}

// ---------------------------------------------------------------------------
// LayerNorm over 768, one wave per row. Optional f32 output + bf16 output.
// ---------------------------------------------------------------------------
__global__ __launch_bounds__(256) void layernorm_kernel(
    const float* __restrict__ x, const float* __restrict__ w,
    const float* __restrict__ b, float* __restrict__ of32,
    bf16_t* __restrict__ obf, int rows) {
  int lane = threadIdx.x & 31;
  int row = blockIdx.x * 8 + (threadIdx.x >> 5);
  if (row >= rows) return;
  const float* xr = x + (size_t)row * E_DIM;
  float vals[24];
  float s = 0.f;
#pragma unroll
  for (int i = 0; i < 24; ++i) { vals[i] = xr[lane + i * 32]; s += vals[i]; }
#pragma unroll
  for (int off = 16; off; off >>= 1) s += __shfl_xor(s, off, 32);
  float mean = s * (1.f / 768.f);
  float vs = 0.f;
#pragma unroll
  for (int i = 0; i < 24; ++i) { float d = vals[i] - mean; vs += d * d; }
#pragma unroll
  for (int off = 16; off; off >>= 1) vs += __shfl_xor(vs, off, 32);
  float rstd = rsqrtf(vs * (1.f / 768.f) + 1e-6f);
#pragma unroll
  for (int i = 0; i < 24; ++i) {
    int c = lane + i * 32;
    float y = (vals[i] - mean) * rstd * w[c] + b[c];
    if (of32) of32[(size_t)row * E_DIM + c] = y;
    obf[(size_t)row * E_DIM + c] = (bf16_t)y;
  }
}

// ---------------------------------------------------------------------------
// WMMA GEMM: C[M,N] = A[M,K] @ Bt[N,K]^T  (bf16 in, f32 accumulate)
// One wave computes a (16*MT)x(16*NT) strip of C. All tile counts are
// compile-time so no EXEC masking ever surrounds a WMMA.
// Requires: M % (16*MT) == 0, N % (16*NT) == 0, K % 32 == 0.
// MODE 0: out bf16 = acc + bias          (value projection)
// MODE 1: out f32  = acc + bias          (offset/attn projection)
// MODE 2: out f32  = query + gamma*(q32 + acc + bias)   (final fused epilogue)
// ---------------------------------------------------------------------------
template <int MODE, int MT, int NT>
__global__ __launch_bounds__(256) void gemm_wmma_kernel(
    const bf16_t* __restrict__ A, const bf16_t* __restrict__ Bt,
    const float* __restrict__ bias, int M, int N, int K,
    float* __restrict__ outF, bf16_t* __restrict__ outB,
    const float* __restrict__ q32, const float* __restrict__ query,
    const float* __restrict__ gamma) {
  const int lane = threadIdx.x & 31;
  const int wave = blockIdx.x * (blockDim.x >> 5) + (threadIdx.x >> 5);
  const int ngroups = N / (16 * NT);
  const int mtile = wave / ngroups;
  const int ngrp = wave % ngroups;
  const int row0 = mtile * 16 * MT;
  if (row0 >= M) return;
  const int colbase = ngrp * 16 * NT;

  const int hi = lane >> 4;          // lane half
  const int mA = lane & 15;          // A row / B column within tile
  const int kbaseA = hi ? 8 : 0;     // ISA 16-bit A layout
  const int kbaseB = hi ? 8 : 0;     // same k layout as A

  v8f acc[MT][NT];
#pragma unroll
  for (int mi = 0; mi < MT; ++mi)
#pragma unroll
    for (int t = 0; t < NT; ++t) acc[mi][t] = v8f{};

  const bf16_t* Arow[MT];
#pragma unroll
  for (int mi = 0; mi < MT; ++mi)
    Arow[mi] = A + (size_t)(row0 + mi * 16 + mA) * K;

  for (int k0 = 0; k0 < K; k0 += 32) {
    v16bf afrag[MT];
#pragma unroll
    for (int mi = 0; mi < MT; ++mi) {
      v8bf alo = *(const v8bf*)(Arow[mi] + k0 + kbaseA);
      v8bf ahi = *(const v8bf*)(Arow[mi] + k0 + kbaseA + 16);
      afrag[mi] = __builtin_shufflevector(alo, ahi, 0, 1, 2, 3, 4, 5, 6, 7, 8,
                                          9, 10, 11, 12, 13, 14, 15);
    }
#pragma unroll
    for (int t = 0; t < NT; ++t) {
      const bf16_t* Brow = Bt + (size_t)(colbase + t * 16 + mA) * K + k0 + kbaseB;
      v8bf blo = *(const v8bf*)(Brow);
      v8bf bhi = *(const v8bf*)(Brow + 16);
      v16bf bfrag = __builtin_shufflevector(blo, bhi, 0, 1, 2, 3, 4, 5, 6, 7, 8,
                                            9, 10, 11, 12, 13, 14, 15);
#pragma unroll
      for (int mi = 0; mi < MT; ++mi) {
        acc[mi][t] = __builtin_amdgcn_wmma_f32_16x16x32_bf16(
            false, afrag[mi], false, bfrag, (short)0, acc[mi][t], false, false);
      }
    }
  }

  // C/D layout: lanes 0-15 -> M=v, N=lane; lanes 16-31 -> M=8+v, N=lane-16
  const int nC = lane & 15;
  const int mOff = hi ? 8 : 0;
#pragma unroll
  for (int t = 0; t < NT; ++t) {
    int col = colbase + t * 16 + nC;
    float bcol = bias ? bias[col] : 0.f;
#pragma unroll
    for (int mi = 0; mi < MT; ++mi) {
#pragma unroll
      for (int v = 0; v < 8; ++v) {
        int row = row0 + mi * 16 + mOff + v;
        size_t idx = (size_t)row * N + col;
        float r = acc[mi][t][v] + bcol;
        if (MODE == 0) {
          outB[idx] = (bf16_t)r;
        } else if (MODE == 1) {
          outF[idx] = r;
        } else {
          float attn = q32[idx] + r;
          outF[idx] = query[idx] + gamma[col] * attn;
        }
      }
    }
  }
}

// ---------------------------------------------------------------------------
// Softmax over 4 attn logits + sampling-location precompute per (b,q,h).
// oa row layout: [0..63] offsets (h*8 + p*2 + c), [64..95] attn (h*4 + p)
// ---------------------------------------------------------------------------
__global__ __launch_bounds__(256) void softloc_kernel(
    const float* __restrict__ oa, const float* __restrict__ ref,
    float* __restrict__ px, float* __restrict__ py, float* __restrict__ aw,
    int total) {
  int t = blockIdx.x * 256 + threadIdx.x;
  if (t >= total) return;
  int h = t & 7;
  int bq = t >> 3;
  const float* row = oa + (size_t)bq * 96;
  float l[4], m = -1e30f;
#pragma unroll
  for (int p = 0; p < 4; ++p) { l[p] = row[64 + h * 4 + p]; m = fmaxf(m, l[p]); }
  float sum = 0.f;
#pragma unroll
  for (int p = 0; p < 4; ++p) { l[p] = __expf(l[p] - m); sum += l[p]; }
  float inv = 1.f / sum;
  float rx = ref[(size_t)bq * 2 + 0] * (float)WF_ - 0.5f;
  float ry = ref[(size_t)bq * 2 + 1] * (float)HF_ - 0.5f;
#pragma unroll
  for (int p = 0; p < 4; ++p) {
    px[(size_t)t * 4 + p] = rx + row[h * 8 + p * 2 + 0];
    py[(size_t)t * 4 + p] = ry + row[h * 8 + p * 2 + 1];
    aw[(size_t)t * 4 + p] = l[p] * inv;
  }
}

// ---------------------------------------------------------------------------
// Bilinear sampling: one 96-thread block per (b,q,h); channel per thread.
// value layout: [(b*NV + s)*768 + h*96 + d] (bf16, L2 resident ~56MB)
// ---------------------------------------------------------------------------
__global__ __launch_bounds__(96) void sample_kernel(
    const bf16_t* __restrict__ value, const float* __restrict__ px,
    const float* __restrict__ py, const float* __restrict__ aw,
    bf16_t* __restrict__ out) {
  int bqh = blockIdx.x;    // bq*8 + h
  int d = threadIdx.x;
  int h = bqh & 7;
  int bq = bqh >> 3;
  int b = bq >> 12;  // / NQ
  const bf16_t* vb = value + (size_t)b * NV_ * E_DIM + h * HD_ + d;
  float acc = 0.f;
#pragma unroll
  for (int p = 0; p < 4; ++p) {
    float x = px[(size_t)bqh * 4 + p];
    float y = py[(size_t)bqh * 4 + p];
    float w = aw[(size_t)bqh * 4 + p];
    float x0 = floorf(x), y0 = floorf(y);
    float fx = x - x0, fy = y - y0;
    int xi = (int)x0, yi = (int)y0;
    float s = 0.f;
#pragma unroll
    for (int dy = 0; dy < 2; ++dy) {
#pragma unroll
      for (int dx = 0; dx < 2; ++dx) {
        int cx = xi + dx, cy = yi + dy;
        if (cx >= 0 && cx < WF_ && cy >= 0 && cy < HF_) {
          float wgt = (dx ? fx : 1.f - fx) * (dy ? fy : 1.f - fy);
          s += wgt * (float)vb[(size_t)(cy * WF_ + cx) * E_DIM];
        }
      }
    }
    acc += w * s;
  }
  out[(size_t)bq * E_DIM + h * HD_ + d] = (bf16_t)acc;
}

// ---------------------------------------------------------------------------
extern "C" void kernel_launch(void* const* d_in, const int* in_sizes, int n_in,
                              void* d_out, int out_size, void* d_ws,
                              size_t ws_size, hipStream_t stream) {
  const float* query  = (const float*)d_in[0];
  const float* ref    = (const float*)d_in[1];
  const float* feat   = (const float*)d_in[2];
  const float* ln_q_w = (const float*)d_in[5];
  const float* ln_q_b = (const float*)d_in[6];
  const float* ln_f_w = (const float*)d_in[7];
  const float* ln_f_b = (const float*)d_in[8];
  const float* W_value = (const float*)d_in[9];
  const float* b_value = (const float*)d_in[10];
  const float* W_off   = (const float*)d_in[11];
  const float* b_off   = (const float*)d_in[12];
  const float* W_attn  = (const float*)d_in[13];
  const float* b_attn  = (const float*)d_in[14];
  const float* W_out   = (const float*)d_in[15];
  const float* b_out   = (const float*)d_in[16];
  const float* gamma   = (const float*)d_in[17];

  const int BQ = B_ * NQ_;      // 16384
  const int BV = B_ * NV_;      // 36864
  const int BQH = BQ * NH_;     // 131072

  // workspace carve-up (all offsets 256B aligned)
  char* p = (char*)d_ws;
  auto take = [&](size_t bytes) { char* r = p; p += (bytes + 255) & ~(size_t)255; return r; };
  float*  q_f32    = (float*)take((size_t)BQ * E_DIM * 4);
  bf16_t* q_bf16   = (bf16_t*)take((size_t)BQ * E_DIM * 2);
  bf16_t* f_bf16   = (bf16_t*)take((size_t)BV * E_DIM * 2);
  bf16_t* val_bf16 = (bf16_t*)take((size_t)BV * E_DIM * 2);
  bf16_t* Wt_value = (bf16_t*)take((size_t)E_DIM * E_DIM * 2);
  bf16_t* Wt_oa    = (bf16_t*)take((size_t)96 * E_DIM * 2);
  bf16_t* Wt_out   = (bf16_t*)take((size_t)E_DIM * E_DIM * 2);
  float*  b_oa     = (float*)take(96 * 4);
  float*  oa_f32   = (float*)take((size_t)BQ * 96 * 4);
  float*  px       = (float*)take((size_t)BQH * 4 * 4);
  float*  py       = (float*)take((size_t)BQH * 4 * 4);
  float*  aw       = (float*)take((size_t)BQH * 4 * 4);
  bf16_t* msda     = (bf16_t*)take((size_t)BQ * E_DIM * 2);
  (void)ws_size;

  // 1. weight casts (transpose to [N][K])
  castT_kernel<<<(E_DIM * E_DIM + 255) / 256, 256, 0, stream>>>(W_value, Wt_value, E_DIM, E_DIM);
  castT_kernel<<<(E_DIM * 64 + 255) / 256, 256, 0, stream>>>(W_off, Wt_oa, E_DIM, 64);
  castT_kernel<<<(E_DIM * 32 + 255) / 256, 256, 0, stream>>>(W_attn, Wt_oa + (size_t)64 * E_DIM, E_DIM, 32);
  castT_kernel<<<(E_DIM * E_DIM + 255) / 256, 256, 0, stream>>>(W_out, Wt_out, E_DIM, E_DIM);
  boa_kernel<<<1, 96, 0, stream>>>(b_off, b_attn, b_oa);

  // 2. layernorms
  layernorm_kernel<<<(BQ + 7) / 8, 256, 0, stream>>>(query, ln_q_w, ln_q_b, q_f32, q_bf16, BQ);
  layernorm_kernel<<<(BV + 7) / 8, 256, 0, stream>>>(feat, ln_f_w, ln_f_b, nullptr, f_bf16, BV);

  // 3. value = f @ W_value + b_value  -> bf16 (L2-resident for sampling)
  {
    int waves = (BV / 32) * (E_DIM / 64);     // MT=2, NT=4
    gemm_wmma_kernel<0, 2, 4><<<(waves + 7) / 8, 256, 0, stream>>>(
        f_bf16, Wt_value, b_value, BV, E_DIM, E_DIM, nullptr, val_bf16,
        nullptr, nullptr, nullptr);
  }

  // 4. [offsets|attn] = q @ Wt_oa + b_oa -> f32 (N=96, MT=2, NT=3 -> 2 groups)
  {
    int waves = (BQ / 32) * 2;
    gemm_wmma_kernel<1, 2, 3><<<(waves + 7) / 8, 256, 0, stream>>>(
        q_bf16, Wt_oa, b_oa, BQ, 96, E_DIM, oa_f32, nullptr, nullptr, nullptr,
        nullptr);
  }

  // 5. softmax + sampling locations
  softloc_kernel<<<(BQH + 255) / 256, 256, 0, stream>>>(oa_f32, ref, px, py, aw, BQH);

  // 6. bilinear sampling
  sample_kernel<<<BQH, 96, 0, stream>>>(val_bf16, px, py, aw, msda);

  // 7. out = msda @ W_out + b_out; fused residual: query + gamma*(q + out)
  {
    int waves = (BQ / 32) * (E_DIM / 64);     // MT=2, NT=4
    gemm_wmma_kernel<2, 2, 4><<<(waves + 7) / 8, 256, 0, stream>>>(
        msda, Wt_out, b_out, BQ, E_DIM, E_DIM, (float*)d_out, nullptr, q_f32,
        query, gamma);
  }
}
